// LucyRNN_41034117546124
// MI455X (gfx1250) — compile-verified, matrix operand order checked by reading, not run on hardware
//
#include <hip/hip_runtime.h>
#include <hip/hip_bf16.h>

// ---------------------------------------------------------------------------
// LucyRNN on MI455X (gfx1250): bf16 WMMA GEMMs with async global->LDS double
// buffering + fused LN / scans.
// B=8, T=1024, DIN=D=1024, L=4, V=8192, M=B*T=8192.
// ---------------------------------------------------------------------------

typedef __attribute__((ext_vector_type(16))) __bf16 v16bf;
typedef __attribute__((ext_vector_type(8)))  float  v8f;
typedef __attribute__((ext_vector_type(4)))  int    v4i;

#define DIM    1024
#define MROWS  8192
#define VOUT   8192
#define NLAYER 4
#define BDIM   8
#define TDIM   1024

#if defined(__gfx1250__) &&                                            \
    __has_builtin(__builtin_amdgcn_global_load_async_to_lds_b128) &&   \
    __has_builtin(__builtin_amdgcn_s_wait_asynccnt)
#define USE_ASYNC_LDS 1
#else
#define USE_ASYNC_LDS 0
#endif

static __device__ __forceinline__ __bf16 f2bf(float f) {
    // round-to-nearest-even f32 -> bf16 via integer math
    unsigned u = __builtin_bit_cast(unsigned, f);
    unsigned r = (u + 0x7FFFu + ((u >> 16) & 1u)) >> 16;
    unsigned short h = (unsigned short)r;
    return __builtin_bit_cast(__bf16, h);
}

#if USE_ASYNC_LDS
// AS1 pointer value is bit-identical to the generic global address.
static __device__ __forceinline__ __attribute__((address_space(1))) v4i*
gbl_as1(const void* p) {
    return (__attribute__((address_space(1))) v4i*)(unsigned long long)p;
}
// AS3 pointer = low 32 bits of generic LDS address (ISA 10.2).
static __device__ __forceinline__ __attribute__((address_space(3))) v4i*
lds_as3(void* p) {
    return (__attribute__((address_space(3))) v4i*)(unsigned)(unsigned long long)p;
}
#endif

// ---------------------------------------------------------------------------
// f32 -> bf16 cast
// ---------------------------------------------------------------------------
__global__ void cast_f32_bf16_kernel(const float* __restrict__ src,
                                     __bf16* __restrict__ dst, size_t n) {
    size_t i = (size_t)blockIdx.x * blockDim.x + threadIdx.x;
    if (i < n) dst[i] = f2bf(src[i]);
}

// ---------------------------------------------------------------------------
// GEMM: C[M,N] = A[M,K] @ W[N,K]^T + bias[N]
// A, W bf16 (row-major, K contiguous for both -> natural WMMA A/B fragments).
// Block tile 128x128x32; 8 waves, each wave 64x32 (4x2 WMMA 16x16 tiles).
// Double-buffered LDS fed by GLOBAL_LOAD_ASYNC_TO_LDS_B128 when available.
// ---------------------------------------------------------------------------
#define TM 128
#define TN 128
#define TK 32
#define LDP (TK + 8)   // pad 8 bf16 = 16B: keeps 16B alignment, breaks bank repeat

__global__ __launch_bounds__(256)
void gemm_bf16_wmma_kernel(const __bf16* __restrict__ A,
                           const __bf16* __restrict__ W,
                           const float*  __restrict__ bias,
                           float* __restrict__ C,
                           int M, int N, int K) {
    __shared__ __bf16 As[2][TM][LDP];
    __shared__ __bf16 Bs[2][TN][LDP];

    const int m0   = blockIdx.y * TM;
    const int n0   = blockIdx.x * TN;
    const int tid  = threadIdx.x;
    const int lane = tid & 31;
    const int wid  = tid >> 5;          // 0..7
    const int wm   = wid >> 2;          // 0..1  -> 64 rows each
    const int wn   = wid & 3;           // 0..3  -> 32 cols each
    const int row0 = wm * 64;
    const int col0 = wn * 32;
    const int lrow = lane & 15;         // row/col within 16x16 tile
    // ISA per-lane K split for 16-bit A/B fragments:
    // lanes 0-15 hold K in {0..7, 16..23}; lanes 16-31 hold K in {8..15, 24..31}
    const int kh   = (lane >> 4) * 8;

    v8f acc[4][2];
#pragma unroll
    for (int r = 0; r < 4; ++r)
#pragma unroll
        for (int c = 0; c < 2; ++c) acc[r][c] = {};

    // staging: each thread moves 16 contiguous halfs (2x b128) of A and of W
    const int ldr = tid >> 1;           // 0..127
    const int seg = (tid & 1) * 16;     // 0 or 16 halfs within the 32-wide tile
    const __bf16* Ag = A + (size_t)(m0 + ldr) * K + seg;
    const __bf16* Wg = W + (size_t)(n0 + ldr) * K + seg;

    union Frag { uint4 u[2]; v16bf v; };

#define GEMM_COMPUTE(BUF)                                                   \
    {                                                                       \
        Frag fa[4], fb[2];                                                  \
        _Pragma("unroll")                                                   \
        for (int r = 0; r < 4; ++r) {                                       \
            const __bf16* p = &As[BUF][row0 + r * 16 + lrow][kh];           \
            fa[r].u[0] = *(const uint4*)p;                                  \
            fa[r].u[1] = *(const uint4*)(p + 16);                           \
        }                                                                   \
        _Pragma("unroll")                                                   \
        for (int c = 0; c < 2; ++c) {                                       \
            const __bf16* p = &Bs[BUF][col0 + c * 16 + lrow][kh];           \
            fb[c].u[0] = *(const uint4*)p;                                  \
            fb[c].u[1] = *(const uint4*)(p + 16);                           \
        }                                                                   \
        _Pragma("unroll")                                                   \
        for (int r = 0; r < 4; ++r)                                         \
            _Pragma("unroll")                                               \
            for (int c = 0; c < 2; ++c)                                     \
                acc[r][c] = __builtin_amdgcn_wmma_f32_16x16x32_bf16(        \
                    false, fa[r].v, false, fb[c].v,                         \
                    (short)0, acc[r][c], false, false);                     \
    }

#if USE_ASYNC_LDS
    // ---- async global->LDS double-buffered pipeline ----
    // issue 4 b128 async copies per thread per tile (A lo/hi, B lo/hi)
#define GEMM_ISSUE(BUF, K0)                                                 \
    {                                                                       \
        __builtin_amdgcn_global_load_async_to_lds_b128(                     \
            gbl_as1(Ag + (K0)),     lds_as3(&As[BUF][ldr][seg]),     0, 0); \
        __builtin_amdgcn_global_load_async_to_lds_b128(                     \
            gbl_as1(Ag + (K0) + 8), lds_as3(&As[BUF][ldr][seg + 8]), 0, 0); \
        __builtin_amdgcn_global_load_async_to_lds_b128(                     \
            gbl_as1(Wg + (K0)),     lds_as3(&Bs[BUF][ldr][seg]),     0, 0); \
        __builtin_amdgcn_global_load_async_to_lds_b128(                     \
            gbl_as1(Wg + (K0) + 8), lds_as3(&Bs[BUF][ldr][seg + 8]), 0, 0); \
    }

    const int nk = K / TK;
    GEMM_ISSUE(0, 0)
    for (int i = 0; i < nk - 1; ++i) {
        const int buf = i & 1;
        GEMM_ISSUE(buf ^ 1, (i + 1) * TK)
        // own 4 newest copies outstanding -> previous tile's copies complete
        __builtin_amdgcn_s_wait_asynccnt(4);
        __syncthreads();               // all waves' copies for tile i visible
        GEMM_COMPUTE(buf)
        __syncthreads();               // done reading tile i; safe to overwrite
    }
    {
        const int buf = (nk - 1) & 1;
        __builtin_amdgcn_s_wait_asynccnt(0);
        __syncthreads();
        GEMM_COMPUTE(buf)
    }
#else
    // ---- fallback: synchronous global->VGPR->LDS staging ----
    for (int k0 = 0; k0 < K; k0 += TK) {
        __syncthreads();
        const uint4* ap = (const uint4*)(Ag + k0);
        const uint4* wp = (const uint4*)(Wg + k0);
        uint4 a0 = ap[0], a1 = ap[1];
        uint4 b0 = wp[0], b1 = wp[1];
        *(uint4*)&As[0][ldr][seg]     = a0;
        *(uint4*)&As[0][ldr][seg + 8] = a1;
        *(uint4*)&Bs[0][ldr][seg]     = b0;
        *(uint4*)&Bs[0][ldr][seg + 8] = b1;
        __syncthreads();
        GEMM_COMPUTE(0)
    }
#endif

    // C/D layout: VGPR v, lanes 0-15 -> M = v, lanes 16-31 -> M = 8+v; N = lane%16
    const int nlane = lane & 15;
    const int mup   = (lane >> 4) * 8;
#pragma unroll
    for (int r = 0; r < 4; ++r) {
        const int mb = m0 + row0 + r * 16 + mup;
#pragma unroll
        for (int c = 0; c < 2; ++c) {
            const int n = n0 + col0 + c * 16 + nlane;
            const float bb = bias ? bias[n] : 0.0f;
#pragma unroll
            for (int v = 0; v < 8; ++v)
                C[(size_t)(mb + v) * N + n] = acc[r][c][v] + bb;
        }
    }
}

// ---------------------------------------------------------------------------
// LayerNorm over rows of length 1024, optional sigmoid(1)/tanh(2) epilogue.
// One 256-thread block per row; writes f32 and (optionally) bf16.
// ---------------------------------------------------------------------------
__global__ __launch_bounds__(256)
void ln_kernel(const float* __restrict__ X,
               const float* __restrict__ gg,
               const float* __restrict__ bb,
               float*  __restrict__ outF,
               __bf16* __restrict__ outB,
               int mode) {
    const int row = blockIdx.x;
    const float* xr = X + (size_t)row * DIM;
    float v[4];
    float s1 = 0.0f, s2 = 0.0f;
#pragma unroll
    for (int j = 0; j < 4; ++j) {
        float t = xr[threadIdx.x + j * 256];
        v[j] = t;
        s1 += t;
        s2 += t * t;
    }
#pragma unroll
    for (int off = 16; off > 0; off >>= 1) {
        s1 += __shfl_xor(s1, off, 32);
        s2 += __shfl_xor(s2, off, 32);
    }
    __shared__ float r1[8], r2[8];
    __shared__ float mu_s, rs_s;
    const int wv = threadIdx.x >> 5;
    if ((threadIdx.x & 31) == 0) { r1[wv] = s1; r2[wv] = s2; }
    __syncthreads();
    if (threadIdx.x == 0) {
        float a = 0.0f, q = 0.0f;
#pragma unroll
        for (int i = 0; i < 8; ++i) { a += r1[i]; q += r2[i]; }
        float mu  = a * (1.0f / DIM);
        float var = q * (1.0f / DIM) - mu * mu;
        mu_s = mu;
        rs_s = rsqrtf(var + 1e-5f);
    }
    __syncthreads();
    const float mu = mu_s, rs = rs_s;
#pragma unroll
    for (int j = 0; j < 4; ++j) {
        const int i = threadIdx.x + j * 256;
        float y = (v[j] - mu) * rs * gg[i] + bb[i];
        if (mode == 1)      y = 1.0f / (1.0f + expf(-y));
        else if (mode == 2) y = tanhf(y);
        const size_t idx = (size_t)row * DIM + i;
        outF[idx] = y;
        if (outB) outB[idx] = f2bf(y);
    }
}

// ---------------------------------------------------------------------------
// Elementwise: decay = sigmoid(dpre) in place; kv = k*v into k buffer
// ---------------------------------------------------------------------------
__global__ void ew_sig_mul_kernel(float* __restrict__ dpre,
                                  float* __restrict__ kbuf,
                                  const float* __restrict__ vbuf, size_t n) {
    size_t i = (size_t)blockIdx.x * blockDim.x + threadIdx.x;
    if (i < n) {
        dpre[i] = 1.0f / (1.0f + expf(-dpre[i]));
        kbuf[i] = kbuf[i] * vbuf[i];
    }
}

// ---------------------------------------------------------------------------
// Scan 1: s_t = a_t*s_{t-1} + kv_t ; s_cell = a_t*s_t + kv_t ;
//         hb = bf16(u + s_cell).  One thread per (b,d) channel.
// ---------------------------------------------------------------------------
__global__ void scan_s_kernel(const float* __restrict__ decay,
                              const float* __restrict__ kv,
                              const float* __restrict__ u,
                              __bf16* __restrict__ hb) {
    const int ch = blockIdx.x * blockDim.x + threadIdx.x;   // 0..B*D-1
    const int b  = ch >> 10;
    const int d  = ch & (DIM - 1);
    size_t idx = (size_t)b * TDIM * DIM + d;
    float s = 0.0f;
    for (int t = 0; t < TDIM; ++t, idx += DIM) {
        const float a  = decay[idx];
        const float kb = kv[idx];
        s = fmaf(a, s, kb);
        const float sc = fmaf(a, s, kb);
        hb[idx] = f2bf(u[idx] + sc);
    }
}

// ---------------------------------------------------------------------------
// Scan 2: h_t = z_t*h_{t-1} + (1-z_t)*c_t ; xb = bf16(h)
// ---------------------------------------------------------------------------
__global__ void scan_h_kernel(const float* __restrict__ z,
                              const float* __restrict__ c,
                              __bf16* __restrict__ xb) {
    const int ch = blockIdx.x * blockDim.x + threadIdx.x;
    const int b  = ch >> 10;
    const int d  = ch & (DIM - 1);
    size_t idx = (size_t)b * TDIM * DIM + d;
    float h = 0.0f;
    for (int t = 0; t < TDIM; ++t, idx += DIM) {
        const float zz = z[idx];
        h = zz * h + (1.0f - zz) * c[idx];
        xb[idx] = f2bf(h);
    }
}

// ---------------------------------------------------------------------------
// Host orchestration
// ---------------------------------------------------------------------------
extern "C" void kernel_launch(void* const* d_in, const int* in_sizes, int n_in,
                              void* d_out, int out_size, void* d_ws, size_t ws_size,
                              hipStream_t stream) {
    (void)in_sizes; (void)n_in; (void)out_size; (void)ws_size;

    // ---- gather inputs (setup_inputs dict order) ----
    int p = 0;
    const float* x = (const float*)d_in[p++];
    const float *Wf[NLAYER][6];   // Wp, Wz, Wk, Wv, Wh, Wd
    const float *bf_[NLAYER][6];  // bp, bz, bk, bv, bh, bd
    const float *lng[NLAYER][3], *lnb[NLAYER][3]; // in, z, h
    for (int l = 0; l < NLAYER; ++l) {
        Wf[l][0] = (const float*)d_in[p++]; bf_[l][0] = (const float*)d_in[p++]; // Wp,bp
        Wf[l][1] = (const float*)d_in[p++]; bf_[l][1] = (const float*)d_in[p++]; // Wz,bz
        Wf[l][2] = (const float*)d_in[p++]; bf_[l][2] = (const float*)d_in[p++]; // Wk,bk
        Wf[l][3] = (const float*)d_in[p++]; bf_[l][3] = (const float*)d_in[p++]; // Wv,bv
        Wf[l][4] = (const float*)d_in[p++]; bf_[l][4] = (const float*)d_in[p++]; // Wh,bh
        Wf[l][5] = (const float*)d_in[p++]; bf_[l][5] = (const float*)d_in[p++]; // Wd,bd
        lng[l][0] = (const float*)d_in[p++]; lnb[l][0] = (const float*)d_in[p++];
        lng[l][1] = (const float*)d_in[p++]; lnb[l][1] = (const float*)d_in[p++];
        lng[l][2] = (const float*)d_in[p++]; lnb[l][2] = (const float*)d_in[p++];
    }
    const float* Wo = (const float*)d_in[p++];
    const float* bo = (const float*)d_in[p++];

    // ---- carve workspace ----
    char* ws = (char*)d_ws;
    auto carve = [&](size_t bytes) { void* r = (void*)ws; ws += bytes; return r; };
    const size_t MD = (size_t)MROWS * DIM;

    __bf16* Wb[NLAYER][6];
    for (int l = 0; l < NLAYER; ++l)
        for (int i = 0; i < 6; ++i)
            Wb[l][i] = (__bf16*)carve((size_t)DIM * DIM * sizeof(__bf16));
    __bf16* Wob = (__bf16*)carve((size_t)VOUT * DIM * sizeof(__bf16));
    __bf16* xb  = (__bf16*)carve(MD * sizeof(__bf16));
    __bf16* ub  = (__bf16*)carve(MD * sizeof(__bf16));
    __bf16* hb  = (__bf16*)carve(MD * sizeof(__bf16));
    float* u32  = (float*)carve(MD * sizeof(float));
    float* t0   = (float*)carve(MD * sizeof(float));
    float* t1   = (float*)carve(MD * sizeof(float));
    float* t2   = (float*)carve(MD * sizeof(float));
    float* t3   = (float*)carve(MD * sizeof(float));

    auto cast = [&](const float* s, __bf16* d, size_t n) {
        cast_f32_bf16_kernel<<<(unsigned)((n + 255) / 256), 256, 0, stream>>>(s, d, n);
    };
    auto gemm = [&](const __bf16* A, const __bf16* W, const float* bias,
                    float* C, int M, int N, int K) {
        dim3 grid(N / TN, M / TM);
        gemm_bf16_wmma_kernel<<<grid, 256, 0, stream>>>(A, W, bias, C, M, N, K);
    };
    auto lnorm = [&](const float* X, const float* g, const float* b,
                     float* oF, __bf16* oB, int mode) {
        ln_kernel<<<MROWS, 256, 0, stream>>>(X, g, b, oF, oB, mode);
    };

    // ---- weight + input casts ----
    for (int l = 0; l < NLAYER; ++l)
        for (int i = 0; i < 6; ++i)
            cast(Wf[l][i], Wb[l][i], (size_t)DIM * DIM);
    cast(Wo, Wob, (size_t)VOUT * DIM);
    cast(x, xb, MD);

    const int nch = BDIM * DIM;  // scan channels

    // ---- layers ----
    for (int l = 0; l < NLAYER; ++l) {
        // u = LN(x @ Wp^T + bp)
        gemm(xb, Wb[l][0], bf_[l][0], t0, MROWS, DIM, DIM);
        lnorm(t0, lng[l][0], lnb[l][0], u32, ub, 0);

        // pre-activations
        gemm(ub, Wb[l][1], bf_[l][1], t0, MROWS, DIM, DIM);  // z pre
        gemm(ub, Wb[l][2], bf_[l][2], t1, MROWS, DIM, DIM);  // k
        gemm(ub, Wb[l][3], bf_[l][3], t2, MROWS, DIM, DIM);  // v
        gemm(ub, Wb[l][5], bf_[l][5], t3, MROWS, DIM, DIM);  // decay pre

        // z = sigmoid(LN(zpre)); decay = sigmoid(dpre); kv = k*v
        lnorm(t0, lng[l][1], lnb[l][1], t0, nullptr, 1);
        ew_sig_mul_kernel<<<(unsigned)((MD + 255) / 256), 256, 0, stream>>>(t3, t1, t2, MD);

        // s-scan fused with s_cell and (u + s_cell) -> bf16
        scan_s_kernel<<<nch / 256, 256, 0, stream>>>(t3, t1, u32, hb);

        // c = tanh(LN((u+s_cell) @ Wh^T + bh))
        gemm(hb, Wb[l][4], bf_[l][4], t2, MROWS, DIM, DIM);
        lnorm(t2, lng[l][2], lnb[l][2], t2, nullptr, 2);

        // h-scan -> bf16 input for next layer
        scan_h_kernel<<<nch / 256, 256, 0, stream>>>(t0, t2, xb);
    }

    // ---- output projection ----
    gemm(xb, Wob, bo, (float*)d_out, MROWS, VOUT, DIM);
}
